// Uni_29016799052628
// MI455X (gfx1250) — matrix operand order repaired and checked by reading, not compiled, not run on hardware
//
#include <hip/hip_runtime.h>
#include <hip/hip_bf16.h>

typedef __attribute__((ext_vector_type(16))) _Float16 v16h;
typedef __attribute__((ext_vector_type(8)))  float    v8f;

#define TPB 256

// ---------------------------------------------------------------- edge packing
// int64 -> int2 once; all subsequent edge passes read 8B/edge instead of 16B.
__global__ void k_pack_edges(const long long* __restrict__ row, const long long* __restrict__ col,
                             int2* __restrict__ rc, int E) {
  int e = blockIdx.x * blockDim.x + threadIdx.x;
  if (e < E) rc[e] = make_int2((int)row[e], (int)col[e]);
}

// ---------------------------------------------------------------- degree / norm
__global__ void k_deg_init(float* __restrict__ deg, int N) {
  int i = blockIdx.x * blockDim.x + threadIdx.x;
  if (i < N) deg[i] = 1.0f;                       // self-loop weight (fill_value=1)
}

__global__ void k_deg_edges(const int2* __restrict__ rc, const float* __restrict__ w,
                            float* __restrict__ deg, int E) {
  int e = blockIdx.x * blockDim.x + threadIdx.x;
  if (e < E) atomicAdd(&deg[rc[e].y], w[e]);
}

__global__ void k_dis(const float* __restrict__ deg,
                      float* __restrict__ dis, float* __restrict__ selfnorm, int N) {
  int i = blockIdx.x * blockDim.x + threadIdx.x;
  if (i >= N) return;
  float d = deg[i];
  float s = (d > 0.0f) ? rsqrtf(fmaxf(d, 1e-30f)) : 0.0f;
  dis[i] = s;
  selfnorm[i] = s * s;                            // norm of self-loop edge (i,i)
}

__global__ void k_norm(const int2* __restrict__ rc, const float* __restrict__ w,
                       const float* __restrict__ dis, float* __restrict__ norm, int E) {
  int e = blockIdx.x * blockDim.x + threadIdx.x;
  if (e >= E) return;
  int2 p = rc[e];
  norm[e] = dis[p.x] * w[e] * dis[p.y];
}

// --------------------------------------------- truncated expm of skew(W) -> f16
// One workgroup (256 thr) per layer; S, term in LDS; each thread owns 16 elems.
// Output is stored PRE-SWIZZLED into the WMMA B-fragment layout:
//   element j of lane (hs,n) for (kchunk,ct)  <=  B[kchunk*32 + (j<8 ? hs*8 + j : 16 + hs*8 + j-8)][ct*16 + n]
//   linear index: ((kchunk*4 + ct)*32 + lane)*16 + j
// so the matmul kernel loads each fragment as one contiguous v16h (2x b128).
__global__ void k_taylor_orth(const float* __restrict__ W_orth, _Float16* __restrict__ orthB) {
  __shared__ float S[4096];
  __shared__ float term[4096];
  const float* W = W_orth + (size_t)blockIdx.x * 4096;
  float accl[16], ntl[16];
  int base = threadIdx.x * 16;
#pragma unroll
  for (int j = 0; j < 16; ++j) {
    int idx = base + j, r = idx >> 6, c = idx & 63;
    S[idx] = W[r * 64 + c] - W[c * 64 + r];       // skew-symmetric part
    float id = (r == c) ? 1.0f : 0.0f;
    term[idx] = id; accl[j] = id;
  }
  __syncthreads();
  for (int k = 1; k <= 10; ++k) {                 // T_TAYLOR = 10
    float invk = 1.0f / (float)k;
#pragma unroll
    for (int j = 0; j < 16; ++j) {
      int idx = base + j, r = idx >> 6, c = idx & 63;
      float s = 0.0f;
      for (int jj = 0; jj < 64; ++jj) s += term[(r << 6) + jj] * S[(jj << 6) + c];
      ntl[j] = s * invk;
    }
    __syncthreads();
#pragma unroll
    for (int j = 0; j < 16; ++j) { int idx = base + j; term[idx] = ntl[j]; accl[j] += ntl[j]; }
    __syncthreads();
  }
#pragma unroll
  for (int j = 0; j < 16; ++j) {
    int idx = base + j, r = idx >> 6, c = idx & 63;        // r = K row, c = N col
    int kchunk = r >> 5, rem = r & 31;
    int hs = (rem >> 3) & 1, jj = (rem & 7) + ((rem >> 4) << 3);
    int ct = c >> 4,  nn = c & 15;
    int lane = hs * 16 + nn;
    orthB[(size_t)blockIdx.x * 4096 + (size_t)((kchunk * 4 + ct) * 32 + lane) * 16 + jj] =
        (_Float16)accl[j];
  }
}

// ---------------------------------------------------------------- 1-wide propagate
__global__ void k_prop1_init(const float* __restrict__ selfnorm, const float* __restrict__ xin,
                             float* __restrict__ p, int N, const float* __restrict__ bias) {
  int i = blockIdx.x * blockDim.x + threadIdx.x;
  if (i >= N) return;
  float b = bias ? bias[0] : 0.0f;
  p[i] = selfnorm[i] * xin[i] + b;
}

__global__ void k_prop1_edges(const int2* __restrict__ rc, const float* __restrict__ norm,
                              const float* __restrict__ xin, float* __restrict__ p, int E) {
  int e = blockIdx.x * blockDim.x + threadIdx.x;
  if (e >= E) return;
  int2 pr = rc[e];
  atomicAdd(&p[pr.y], norm[e] * xin[pr.x]);
}

// h[i][j] = p[i]*W0[j] + b0[j]; also emits tmp = selfnorm (.) h (propagate init fused)
__global__ void k_expand(const float* __restrict__ p, const float* __restrict__ W0,
                         const float* __restrict__ b0, float* __restrict__ h,
                         float* __restrict__ tmp, const float* __restrict__ selfnorm, int N) {
  int t = blockIdx.x * blockDim.x + threadIdx.x;
  int i = t >> 6, j = t & 63;
  if (i >= N) return;
  float v = p[i] * W0[j] + b0[j];
  h[t] = v;
  tmp[t] = selfnorm[i] * v;
}

// q[i] = dot(h[i,:], W11)   (64->1 contraction before the final propagate)
__global__ void k_dotW11(const float* __restrict__ h, const float* __restrict__ W11,
                         float* __restrict__ q, int N) {
  int i = blockIdx.x * blockDim.x + threadIdx.x;
  if (i >= N) return;
  const float4* hp = (const float4*)(h + (size_t)i * 64);
  const float4* wp = (const float4*)W11;
  float s = 0.0f;
#pragma unroll
  for (int j = 0; j < 16; ++j) {
    float4 a = hp[j], b = wp[j];
    s += a.x * b.x + a.y * b.y + a.z * b.z + a.w * b.w;
  }
  q[i] = s;
}

// ---------------------------------------------------------------- 64-wide propagate
// 16 threads per edge, float4 gather per thread, scatter via f32 atomics (L2-resident)
__global__ void k_prop64_edges(const int2* __restrict__ rc, const float* __restrict__ norm,
                               const float* __restrict__ hin, float* __restrict__ hout,
                               int E) {
  long long t = (long long)blockIdx.x * blockDim.x + threadIdx.x;
  long long e = t >> 4;
  if (e >= E) return;
  int c = (int)(t & 15) << 2;
  int2 p = rc[e];
  float wv = norm[e];
  float4 v = *(const float4*)(hin + (size_t)p.x * 64 + c);
  float* dst = hout + (size_t)p.y * 64 + c;
  atomicAdd(dst + 0, wv * v.x);
  atomicAdd(dst + 1, wv * v.y);
  atomicAdd(dst + 2, wv * v.z);
  atomicAdd(dst + 3, wv * v.w);
}

// ------------------------------------------------------ h = tmp @ orth  (WMMA f16)
// 256 threads = 8 waves; each wave owns a 16-row node tile and all 64 cols.
// A built per ISA 16-bit A layout from f32 tmp rows; B fragments are single
// contiguous v16h loads (pre-swizzled by k_taylor_orth). Also fuses next
// propagate's init: tmp <- selfnorm (.) h_new (in-place safe: each wave reads
// its full tile into A before storing, tiles disjoint across waves).
__global__ void k_matmul_wmma(float* __restrict__ tmp, const _Float16* __restrict__ orthB,
                              float* __restrict__ hout, const float* __restrict__ selfnorm,
                              int writeTmp, int N) {
  int wave = threadIdx.x >> 5;
  int lane = threadIdx.x & 31;
  int mbase = (blockIdx.x * (blockDim.x >> 5) + wave) * 16;
  if (mbase >= N) return;                          // wave-uniform: EXEC stays all-1s
  int mrow = lane & 15;
  int hs   = lane >> 4;
  int n    = lane & 15;

  // A fragments for K=[0,32) and K=[32,64): 4x 32B contiguous f32 reads + cvt
  const float* rp = tmp + (size_t)(mbase + mrow) * 64;
  v16h a0, a1;
#pragma unroll
  for (int j = 0; j < 8; ++j) {
    a0[j]     = (_Float16)rp[hs * 8 + j];          // K = hs*8 + j
    a0[8 + j] = (_Float16)rp[16 + hs * 8 + j];     // K = 16 + hs*8 + j
    a1[j]     = (_Float16)rp[32 + hs * 8 + j];
    a1[8 + j] = (_Float16)rp[48 + hs * 8 + j];
  }

  const v16h* bp = (const v16h*)orthB;             // 32B-aligned fragments
  v8f acc[4];
#pragma unroll
  for (int ct = 0; ct < 4; ++ct) {
    v16h b0v = bp[(0 * 4 + ct) * 32 + lane];       // K chunk 0
    v16h b1v = bp[(1 * 4 + ct) * 32 + lane];       // K chunk 1
    v8f c = {};
    c = __builtin_amdgcn_wmma_f32_16x16x32_f16(false, a0, false, b0v, (short)0, c, false, false);
    c = __builtin_amdgcn_wmma_f32_16x16x32_f16(false, a1, false, b1v, (short)0, c, false, false);
    acc[ct] = c;
  }

  // D layout: VGPR v -> row mbase + v + 8*hs, col ct*16 + n
  float sn[8];
  if (writeTmp) {
#pragma unroll
    for (int v = 0; v < 8; ++v) sn[v] = selfnorm[mbase + v + 8 * hs];
  }
#pragma unroll
  for (int ct = 0; ct < 4; ++ct) {
#pragma unroll
    for (int v = 0; v < 8; ++v) {
      size_t o = (size_t)(mbase + v + 8 * hs) * 64 + ct * 16 + n;
      float val = acc[ct][v];
      hout[o] = val;
      if (writeTmp) tmp[o] = sn[v] * val;
    }
  }
}

// ------------------------------------------------------------------------- driver
extern "C" void kernel_launch(void* const* d_in, const int* in_sizes, int n_in,
                              void* d_out, int out_size, void* d_ws, size_t ws_size,
                              hipStream_t stream) {
  const float*     x      = (const float*)d_in[0];
  const long long* ei     = (const long long*)d_in[1];   // int64 per reference
  const float*     ew     = (const float*)d_in[2];
  const float*     W0     = (const float*)d_in[3];
  const float*     b0     = (const float*)d_in[4];
  const float*     W_orth = (const float*)d_in[5];
  const float*     W11    = (const float*)d_in[6];
  const float*     b11    = (const float*)d_in[7];

  const int N = in_sizes[0];       // 200000
  const int E = in_sizes[2];       // 1200000
  const long long* row = ei;
  const long long* col = ei + E;

  // workspace carve-up (all buffers fully written before read each call)
  char* ws = (char*)d_ws;
  size_t off = 0;
  auto carve = [&](size_t bytes) -> char* {
    char* p = ws + off; off = (off + bytes + 255) & ~(size_t)255; return p;
  };
  float*     deg      = (float*)carve((size_t)N * 4);
  float*     dis      = (float*)carve((size_t)N * 4);
  float*     selfnorm = (float*)carve((size_t)N * 4);
  float*     norm     = (float*)carve((size_t)E * 4);
  float*     p        = (float*)carve((size_t)N * 4);     // also reused as q
  int2*      rc       = (int2*)carve((size_t)E * 8);
  _Float16*  orthB    = (_Float16*)carve((size_t)10 * 4096 * 2);
  float*     h        = (float*)carve((size_t)N * 64 * 4);
  float*     tmp      = (float*)carve((size_t)N * 64 * 4);
  (void)ws_size; (void)n_in; (void)out_size;

  const int gN   = (N + TPB - 1) / TPB;
  const int gE   = (E + TPB - 1) / TPB;
  const int gN64 = (N * 64 + TPB - 1) / TPB;
  const int gE16 = (int)(((long long)E * 16 + TPB - 1) / TPB);
  const int gMM  = (N + 127) / 128;                       // 8 waves x 16 rows / block

  // edge packing + gcn_norm
  k_pack_edges<<<gE, TPB, 0, stream>>>(row, col, rc, E);
  k_deg_init  <<<gN, TPB, 0, stream>>>(deg, N);
  k_deg_edges <<<gE, TPB, 0, stream>>>(rc, ew, deg, E);
  k_dis       <<<gN, TPB, 0, stream>>>(deg, dis, selfnorm, N);
  k_norm      <<<gE, TPB, 0, stream>>>(rc, ew, dis, norm, E);

  // orthogonal weights: expm_taylor(skew(W_k)) -> f16 in WMMA-B fragment layout
  k_taylor_orth<<<10, TPB, 0, stream>>>(W_orth, orthB);

  // block 0 (rank-1): h = propagate(x) (x) W0 + b0 ; tmp = selfnorm (.) h fused
  k_prop1_init <<<gN, TPB, 0, stream>>>(selfnorm, x, p, N, nullptr);
  k_prop1_edges<<<gE, TPB, 0, stream>>>(rc, norm, x, p, E);
  k_expand     <<<gN64, TPB, 0, stream>>>(p, W0, b0, h, tmp, selfnorm, N);

  // blocks 1..10: h = propagate(h) @ orth_k ; next init fused into matmul
  for (int k = 0; k < 10; ++k) {
    k_prop64_edges<<<gE16, TPB, 0, stream>>>(rc, norm, h, tmp, E);
    k_matmul_wmma <<<gMM, TPB, 0, stream>>>(tmp, orthB + (size_t)k * 4096, h, selfnorm,
                                            (k < 9) ? 1 : 0, N);
  }

  // block 11 (rank-1): out = propagate(h @ W11) + b11
  float* out = (float*)d_out;
  k_dotW11     <<<gN, TPB, 0, stream>>>(h, W11, p, N);
  k_prop1_init <<<gN, TPB, 0, stream>>>(selfnorm, p, out, N, b11);
  k_prop1_edges<<<gE, TPB, 0, stream>>>(rc, norm, p, out, E);
}
